// Biaffine_18940805775917
// MI455X (gfx1250) — compile-verified, matrix operand order checked by reading, not run on hardware
//
#include <hip/hip_runtime.h>
#include <hip/hip_bf16.h>

// Problem constants (from reference): B=4, S=512, D=768, H=200
#define BB   4
#define SS   512
#define DD   768
#define HH   200
#define HP   224          // H padded to multiple of 32 (WMMA K step)
#define BS   (BB * SS)    // 2048 rows

typedef __attribute__((ext_vector_type(16))) _Float16 v16h;
typedef __attribute__((ext_vector_type(8)))  _Float16 v8h;
typedef __attribute__((ext_vector_type(8)))  float    v8f;

// ---------------------------------------------------------------------------
// Fragment loader for v_wmma_f32_16x16x32_f16 (wave32).
// Valid for both A (16xK tile, row-major K-contiguous) and B (KxN tile loaded
// from B^T stored row-major K-contiguous), per ISA 7.12.2 16-bit A layout:
//   lane L: row = row0 + (L & 15), half = L >> 4
//   elements 0..7  -> K = k0 + half*8 + {0..7}
//   elements 8..15 -> K = k0 + 16 + half*8 + {0..7}
// => two global_load_b128 per fragment.
// ---------------------------------------------------------------------------
__device__ __forceinline__ v16h load_frag_k(const _Float16* __restrict__ base,
                                            int ld, int row0, int k0, int lane) {
    const int r    = row0 + (lane & 15);
    const int half = lane >> 4;
    const _Float16* p = base + (size_t)r * ld + k0 + half * 8;
    v8h lo = *(const v8h*)(p);
    v8h hi = *(const v8h*)(p + 16);
    v16h f;
#pragma unroll
    for (int e = 0; e < 8; ++e) { f[e] = lo[e]; f[e + 8] = hi[e]; }
    return f;
}

#define WMMA_F16(A, Bv, Cacc) \
    __builtin_amdgcn_wmma_f32_16x16x32_f16(false, (A), false, (Bv), (short)0, (Cacc), false, false)

// C/D layout (ISA 7.12.2): element r of v8f -> m = r + 8*(lane>>4), n = lane&15
__device__ __forceinline__ void store_tile_f16(const v8f& acc,
                                               _Float16* __restrict__ C, int ldc,
                                               int m0, int n0,
                                               const float* __restrict__ bias,
                                               int nvalid, int do_relu, int lane) {
    const int n  = n0 + (lane & 15);
    const int mb = m0 + ((lane >> 4) << 3);
    const float bv = (bias != nullptr && n < nvalid) ? bias[n] : 0.0f;
#pragma unroll
    for (int r = 0; r < 8; ++r) {
        float v = acc[r] + bv;
        if (do_relu) v = v > 0.0f ? v : 0.0f;
        if (n >= nvalid) v = 0.0f;
        C[(size_t)(mb + r) * ldc + n] = (_Float16)v;
    }
}

__device__ __forceinline__ void store_tile_f32(const v8f& acc,
                                               float* __restrict__ O, int ldo,
                                               int m0, int n0,
                                               float addv, float scale, int lane) {
    const int n  = n0 + (lane & 15);
    const int mb = m0 + ((lane >> 4) << 3);
#pragma unroll
    for (int r = 0; r < 8; ++r)
        O[(size_t)(mb + r) * ldo + n] = (acc[r] + addv) * scale;
}

// ---------------------------------------------------------------------------
// Elementwise f32 -> f16 convert (for x)
// ---------------------------------------------------------------------------
__global__ void cvt_f32_to_f16(const float* __restrict__ in,
                               _Float16* __restrict__ out, int n) {
    int t = blockIdx.x * blockDim.x + threadIdx.x;
    if (t < n) out[t] = (_Float16)in[t];
}

// ---------------------------------------------------------------------------
// W [D,H] (f32) -> W^T [Hp,D] (f16), zero-padded rows h >= H
// ---------------------------------------------------------------------------
__global__ void transpose_w_f16(const float* __restrict__ W,
                                _Float16* __restrict__ WT) {
    int t = blockIdx.x * blockDim.x + threadIdx.x;
    if (t >= HP * DD) return;
    int d = t % DD;
    int h = t / DD;
    float v = (h < HH) ? W[(size_t)d * HH + h] : 0.0f;
    WT[(size_t)h * DD + d] = (_Float16)v;
}

// ---------------------------------------------------------------------------
// UeffT[j,i] = sum_o U[o,i,j] * W_down[o]   (f16, [Hp x Hp], zero padded)
// ---------------------------------------------------------------------------
__global__ void build_ueffT(const float* __restrict__ U,
                            const float* __restrict__ Wd,
                            _Float16* __restrict__ UeffT) {
    int t = blockIdx.x * blockDim.x + threadIdx.x;
    if (t >= HP * HP) return;
    int j = t % HP;
    int i = t / HP;
    float s = 0.0f;
    if (i < HH && j < HH) {
        const float* up = U + (size_t)i * HH + j;   // stride H*H over o
        for (int o = 0; o < HH; ++o)
            s += up[(size_t)o * HH * HH] * Wd[o];
    }
    UeffT[(size_t)j * HP + i] = (_Float16)s;
}

// ---------------------------------------------------------------------------
// Register-blocked WMMA GEMM: each wave owns a 32x32 output block (2x2 tiles).
// C_f16[m,n] = post( sum_k A[m,k]*BT[n,k] + bias[n] ); cols n>=nvalid forced 0.
// Per K-step: 8 x global_load_b128 -> 4 x v_wmma (2x operand reuse).
// ---------------------------------------------------------------------------
__global__ void gemm_f16_wmma(const _Float16* __restrict__ A, int lda,
                              const _Float16* __restrict__ BT, int ldb,
                              const float* __restrict__ bias, int nvalid,
                              _Float16* __restrict__ C, int ldc,
                              int M, int Np, int K, int do_relu) {
    const int wave = (blockIdx.x * blockDim.x + threadIdx.x) >> 5;
    const int lane = threadIdx.x & 31;
    const int nbn  = Np >> 5;                  // 32-wide N blocks
    const int total = (M >> 5) * nbn;
    if (wave >= total) return;                 // wave-uniform: EXEC stays all-1s
    const int m0 = (wave / nbn) << 5;
    const int n0 = (wave % nbn) << 5;

    v8f acc00 = {}, acc01 = {}, acc10 = {}, acc11 = {};
    for (int k0 = 0; k0 < K; k0 += 32) {
        v16h a0 = load_frag_k(A,  lda, m0,      k0, lane);
        v16h a1 = load_frag_k(A,  lda, m0 + 16, k0, lane);
        v16h b0 = load_frag_k(BT, ldb, n0,      k0, lane);
        v16h b1 = load_frag_k(BT, ldb, n0 + 16, k0, lane);
        acc00 = WMMA_F16(a0, b0, acc00);
        acc01 = WMMA_F16(a0, b1, acc01);
        acc10 = WMMA_F16(a1, b0, acc10);
        acc11 = WMMA_F16(a1, b1, acc11);
    }

    store_tile_f16(acc00, C, ldc, m0,      n0,      bias, nvalid, do_relu, lane);
    store_tile_f16(acc01, C, ldc, m0,      n0 + 16, bias, nvalid, do_relu, lane);
    store_tile_f16(acc10, C, ldc, m0 + 16, n0,      bias, nvalid, do_relu, lane);
    store_tile_f16(acc11, C, ldc, m0 + 16, n0 + 16, bias, nvalid, do_relu, lane);
}

// ---------------------------------------------------------------------------
// Batched score GEMM, 32x32 block per wave:
// out[b,x,y] = (sum_j hU[b,x,j]*tail[b,y,j] + b_down) * scale
// K = HP exact (padded columns of hU/tail are zero).
// ---------------------------------------------------------------------------
__global__ void score_wmma(const _Float16* __restrict__ hU,
                           const _Float16* __restrict__ tail,
                           const float* __restrict__ b_down,
                           float* __restrict__ out, float scale) {
    const int wave = (blockIdx.x * blockDim.x + threadIdx.x) >> 5;
    const int lane = threadIdx.x & 31;
    const int bps  = SS >> 5;                  // 16 blocks per side
    const int bpb  = bps * bps;                // 256 blocks per batch
    const int b  = wave / bpb;
    const int t  = wave % bpb;
    const int x0 = (t / bps) << 5;
    const int y0 = (t % bps) << 5;

    const _Float16* Ab = hU   + (size_t)b * SS * HP;
    const _Float16* Bb = tail + (size_t)b * SS * HP;

    v8f acc00 = {}, acc01 = {}, acc10 = {}, acc11 = {};
#pragma unroll
    for (int k0 = 0; k0 < HP; k0 += 32) {      // 7 iterations
        v16h a0 = load_frag_k(Ab, HP, x0,      k0, lane);
        v16h a1 = load_frag_k(Ab, HP, x0 + 16, k0, lane);
        v16h b0 = load_frag_k(Bb, HP, y0,      k0, lane);
        v16h b1 = load_frag_k(Bb, HP, y0 + 16, k0, lane);
        acc00 = WMMA_F16(a0, b0, acc00);
        acc01 = WMMA_F16(a0, b1, acc01);
        acc10 = WMMA_F16(a1, b0, acc10);
        acc11 = WMMA_F16(a1, b1, acc11);
    }

    const float bd = b_down[0];
    float* o = out + (size_t)b * SS * SS;
    store_tile_f32(acc00, o, SS, x0,      y0,      bd, scale, lane);
    store_tile_f32(acc01, o, SS, x0,      y0 + 16, bd, scale, lane);
    store_tile_f32(acc10, o, SS, x0 + 16, y0,      bd, scale, lane);
    store_tile_f32(acc11, o, SS, x0 + 16, y0 + 16, bd, scale, lane);
}

// ---------------------------------------------------------------------------
// Launcher
// ---------------------------------------------------------------------------
extern "C" void kernel_launch(void* const* d_in, const int* in_sizes, int n_in,
                              void* d_out, int out_size, void* d_ws, size_t ws_size,
                              hipStream_t stream) {
    (void)in_sizes; (void)n_in; (void)out_size; (void)ws_size;

    const float* x      = (const float*)d_in[0];
    const float* W_head = (const float*)d_in[1];
    const float* b_head = (const float*)d_in[2];
    const float* W_tail = (const float*)d_in[3];
    const float* b_tail = (const float*)d_in[4];
    const float* U      = (const float*)d_in[5];
    const float* W_down = (const float*)d_in[6];
    const float* b_down = (const float*)d_in[7];
    float* out = (float*)d_out;

    // Workspace layout (all sizes multiples of 256 bytes -> 16B aligned)
    char* ws = (char*)d_ws;
    _Float16* xF16  = (_Float16*)(ws);                                 // 2048*768
    _Float16* WhT   = (_Float16*)(ws + 3145728);                       // 224*768
    _Float16* WtT   = (_Float16*)(ws + 3145728 + 344064);              // 224*768
    _Float16* headF = (_Float16*)(ws + 3145728 + 2*344064);            // 2048*224
    _Float16* tailF = (_Float16*)(ws + 3145728 + 2*344064 + 917504);   // 2048*224
    _Float16* UeffT = (_Float16*)(ws + 3145728 + 2*344064 + 2*917504); // 224*224
    _Float16* hU    = (_Float16*)(ws + 3145728 + 2*344064 + 2*917504 + 100352);

    const int TB = 256;

    // 1) x -> f16
    {
        int n = BS * DD;
        cvt_f32_to_f16<<<(n + TB - 1) / TB, TB, 0, stream>>>(x, xF16, n);
    }
    // 2) W_head^T, W_tail^T (f16, zero padded rows)
    {
        int n = HP * DD;
        transpose_w_f16<<<(n + TB - 1) / TB, TB, 0, stream>>>(W_head, WhT);
        transpose_w_f16<<<(n + TB - 1) / TB, TB, 0, stream>>>(W_tail, WtT);
    }
    // 3) UeffT[j,i] = sum_o U[o,i,j] * W_down[o]
    {
        int n = HP * HP;
        build_ueffT<<<(n + TB - 1) / TB, TB, 0, stream>>>(U, W_down, UeffT);
    }
    // 4) head/tail = relu(x @ W + b) in f16  (M=2048, Np=224, K=768)
    {
        int waves  = (BS >> 5) * (HP >> 5);        // 64 * 7 = 448 waves
        int blocks = (waves * 32 + TB - 1) / TB;   // 56 blocks
        gemm_f16_wmma<<<blocks, TB, 0, stream>>>(xF16, DD, WhT, DD, b_head, HH,
                                                 headF, HP, BS, HP, DD, 1);
        gemm_f16_wmma<<<blocks, TB, 0, stream>>>(xF16, DD, WtT, DD, b_tail, HH,
                                                 tailF, HP, BS, HP, DD, 1);
    }
    // 5) hU = head @ Ueff  (M=2048, Np=224, K=224); padded cols come out 0
    {
        int waves  = (BS >> 5) * (HP >> 5);
        int blocks = (waves * 32 + TB - 1) / TB;
        gemm_f16_wmma<<<blocks, TB, 0, stream>>>(headF, HP, UeffT, HP, nullptr, HP,
                                                 hU, HP, BS, HP, HP, 0);
    }
    // 6) scores[b,x,y] = (hU[b,x,:].tail[b,y,:] + b_down) / sqrt(200)
    {
        const float scale = 0.070710678118654752f;   // 1/sqrt(200)
        int waves  = BB * (SS >> 5) * (SS >> 5);     // 1024 waves
        int blocks = (waves * 32 + TB - 1) / TB;     // 128 blocks
        score_wmma<<<blocks, TB, 0, stream>>>(hU, tailF, b_down, out, scale);
    }
}
